// TFN_8839042695326
// MI455X (gfx1250) — matrix operand order repaired and checked by reading, not compiled, not run on hardware
//
#include <hip/hip_runtime.h>
#include <hip/hip_fp16.h>
#include <math.h>

// ---------------------------------------------------------------------------
// CDNA5 / gfx1250 TFN point-cloud classifier.
// v_wmma_f32_16x16x32_f16 for both per-target contractions;
// global_load_async_to_lds_b128 (+ s_wait_asynccnt) stages slab weights.
// ---------------------------------------------------------------------------

typedef __attribute__((ext_vector_type(16))) _Float16 v16h;
typedef __attribute__((ext_vector_type(8)))  float    v8f;

#define B_     32
#define N_     2048
#define M_     512
#define C_     64
#define KNN_   32
#define RADIUS_ 0.4f
#define GSCALE_ (0.69314718056f * 9.0f)   // ln2 * NSHELLS^2

// A-matrix (16-bit, 16x32) K index for element j, lane-half h (ISA 7.12.2)
__device__ __forceinline__ int aKmap(int j, int h) { return j + 8 * ((j >> 3) + h); }
// B-matrix (16-bit, 32x16) K index for element j, lane-half h (same K order as A)
__device__ __forceinline__ int bKmap(int j, int h) { return j + 8 * ((j >> 3) + h); }

__device__ __forceinline__ v8f wmma16x16x32(const v16h& a, const v16h& b, const v8f& c) {
  // (neg_a, A, neg_b, B, c_mod, C, reuse_a, reuse_b)
  return __builtin_amdgcn_wmma_f32_16x16x32_f16(false, a, false, b, (short)0, c,
                                                false, false);
}

// Async global -> LDS copy of 16 bytes (per lane), ASYNCcnt-tracked.
__device__ __forceinline__ void async_g2l_b128(unsigned lds_addr, const void* gaddr) {
  asm volatile("global_load_async_to_lds_b128 %0, %1, off"
               :: "v"(lds_addr), "v"((unsigned long long)(uintptr_t)gaddr)
               : "memory");
}
__device__ __forceinline__ void wait_asynccnt0() {
  asm volatile("s_wait_asynccnt 0" ::: "memory");
}

// ---------------------------------------------------------------------------
// Kernel 1: KNN. One wave32 per target; 2048 squared distances in LDS,
// 32 iterative min-extractions via wave32 butterfly reductions.
// ---------------------------------------------------------------------------
__global__ void __launch_bounds__(128) knn_kernel(const float* __restrict__ points,
                                                  int* __restrict__ idxbuf) {
  __shared__ float sd[4][N_];
  const int w    = threadIdx.x >> 5;
  const int lane = threadIdx.x & 31;
  const int t    = blockIdx.x * 4 + w;          // target id 0..16383
  const int b    = t >> 9;
  const int m    = t & (M_ - 1);
  const float* pb = points + (size_t)b * N_ * 3;
  const float tx = pb[(m * 4) * 3 + 0];
  const float ty = pb[(m * 4) * 3 + 1];
  const float tz = pb[(m * 4) * 3 + 2];

  for (int n = lane; n < N_; n += 32) {
    float dx = pb[n * 3 + 0] - tx;
    float dy = pb[n * 3 + 1] - ty;
    float dz = pb[n * 3 + 2] - tz;
    sd[w][n] = dx * dx + dy * dy + dz * dz;
  }
  __syncthreads();

  for (int it = 0; it < KNN_; ++it) {
    float bv = 3.0e38f;
    int   bi = 0x7fffffff;
    for (int n = lane; n < N_; n += 32) {
      float v = sd[w][n];
      if (v < bv || (v == bv && n < bi)) { bv = v; bi = n; }
    }
    // full wave32 butterfly: every lane ends with the global (min, argmin)
    for (int off = 16; off > 0; off >>= 1) {
      float ov = __shfl_xor(bv, off);
      int   oi = __shfl_xor(bi, off);
      if (ov < bv || (ov == bv && oi < bi)) { bv = ov; bi = oi; }
    }
    if (lane == 0) idxbuf[t * KNN_ + it] = bi;
    // owner lane (bi % 32) retires the winner; only that lane re-reads it.
    if ((bi & 31) == lane) sd[w][bi] = 3.0e38f;
  }
}

// ---------------------------------------------------------------------------
// Kernel 2 (fused, WMMA): per target,
//   G(48x32)  = SH(p) * shellW(s) per neighbor k        (LDS, f16)
//   F(32x64)  = gathered features                        (LDS, f16)
//   Z = G @ F (12 wmma)  -> LDS as z[p][s*64+c]
//   per slab l: O = Zslab(16x192 zero-padded) @ Wl(192x64)  (24 wmma)
//     W chunks async-staged into block-shared LDS (8KB at a time)
//   h_all[t, l*64+c] = sqrt(sum_p O^2 + eps)  (bias b0 on slab 0)
// One wave per target, 4 waves per block.
// ---------------------------------------------------------------------------
__global__ void __launch_bounds__(128) tfn_kernel(
    const float* __restrict__ points, const float* __restrict__ feats,
    const int* __restrict__ idxbuf,
    const float* __restrict__ W0, const float* __restrict__ b0,
    const float* __restrict__ W1, const float* __restrict__ W2,
    const float* __restrict__ W3,
    float* __restrict__ h_all) {
  __shared__ __attribute__((aligned(16))) _Float16 sG[4][48 * 32]; // rows=p*3+s, cols=k
  __shared__ __attribute__((aligned(16))) _Float16 sF[4][32 * 64]; // rows=k, cols=c
  __shared__ __attribute__((aligned(16))) _Float16 sZ[4][16 * 192];// z[p][s*64+c]
  __shared__ __attribute__((aligned(16))) float    sW[32 * 64];    // staged W chunk (f32)
  const int w    = threadIdx.x >> 5;
  const int lane = threadIdx.x & 31;
  const int h    = lane >> 4;
  const int ln   = lane & 15;
  const int t    = blockIdx.x * 4 + w;
  const int b    = t >> 9;
  const int m    = t & (M_ - 1);

  // ---- Phase 1: lane k handles neighbor k ----
  {
    const float* pb = points + (size_t)b * N_ * 3;
    const float tx = pb[(m * 4) * 3 + 0];
    const float ty = pb[(m * 4) * 3 + 1];
    const float tz = pb[(m * 4) * 3 + 2];
    const int ni = idxbuf[t * KNN_ + lane];
    const float* fb = feats + ((size_t)b * N_ + ni) * C_;
    __builtin_prefetch(fb, 0, 0);                       // global_prefetch_b8
    const float rx = pb[ni * 3 + 0] - tx;
    const float ry = pb[ni * 3 + 1] - ty;
    const float rz = pb[ni * 3 + 2] - tz;
    const float d2   = rx * rx + ry * ry + rz * rz;
    const float dist = sqrtf(fmaxf(d2, 1e-12f));
    const float inv  = 1.0f / dist;
    const float x = rx * inv, y = ry * inv, z = rz * inv;
    const float x2 = x * x, y2 = y * y, z2 = z * z;
    float Y[16];
    Y[0]  = 0.282095f;
    Y[1]  = 0.488603f * y;
    Y[2]  = 0.488603f * z;
    Y[3]  = 0.488603f * x;
    Y[4]  = 1.092548f * x * y;
    Y[5]  = 1.092548f * y * z;
    Y[6]  = 0.315392f * (3.0f * z2 - 1.0f);
    Y[7]  = 1.092548f * x * z;
    Y[8]  = 0.546274f * (x2 - y2);
    Y[9]  = 0.590044f * y * (3.0f * x2 - y2);
    Y[10] = 2.890611f * x * y * z;
    Y[11] = 0.457046f * y * (5.0f * z2 - 1.0f);
    Y[12] = 0.373176f * z * (5.0f * z2 - 3.0f);
    Y[13] = 0.457046f * x * (5.0f * z2 - 1.0f);
    Y[14] = 1.445306f * z * (x2 - y2);
    Y[15] = 0.590044f * x * (x2 - 3.0f * y2);

    const float dn   = dist / RADIUS_;
    const float mask = (dn <= 1.0f) ? 1.0f : 0.0f;
    float ws_[3];
    float wsum = 0.0f;
    #pragma unroll
    for (int s = 0; s < 3; ++s) {
      const float dd = dn - 0.5f * (float)s;
      ws_[s] = __expf(-GSCALE_ * dd * dd) * mask;
      wsum += ws_[s];
    }
    const float wn = 1.0f / (wsum + 1e-8f);
    #pragma unroll
    for (int p = 0; p < 16; ++p)
      #pragma unroll
      for (int s = 0; s < 3; ++s)
        sG[w][(p * 3 + s) * 32 + lane] = (_Float16)(Y[p] * ws_[s] * wn);

    for (int c = 0; c < C_; ++c) sF[w][lane * C_ + c] = (_Float16)fb[c];
  }
  __syncthreads();

  // ---- Phase 2: Z = G(48x32) @ F(32x64), K=32 in a single wmma per tile ----
  for (int mt = 0; mt < 3; ++mt) {
    v16h a;
    #pragma unroll
    for (int j = 0; j < 16; ++j)
      a[j] = sG[w][(mt * 16 + ln) * 32 + aKmap(j, h)];
    for (int nt = 0; nt < 4; ++nt) {
      v16h bb;
      #pragma unroll
      for (int j = 0; j < 16; ++j)
        bb[j] = sF[w][bKmap(j, h) * 64 + nt * 16 + ln];
      v8f cz = {};
      v8f d = wmma16x16x32(a, bb, cz);
      #pragma unroll
      for (int r = 0; r < 8; ++r) {
        const int row = mt * 16 + r + 8 * h;          // ps index 0..47
        const int p = row / 3, s = row - 3 * p;
        sZ[w][p * 192 + s * 64 + nt * 16 + ln] = (_Float16)d[r];
      }
    }
  }

  // ---- Phase 3: per-slab GEMM (K=192) + norm over degree components ----
  // W chunks (32x64 f32 = 8KB) are shared by all 4 waves: async-stage to LDS.
  const int pstart[4] = {0, 1, 4, 9};
  const int pcnt[4]   = {1, 3, 5, 7};
  const int tid = threadIdx.x;
  for (int l = 0; l < 4; ++l) {
    const float* Wl = (l == 0) ? W0 : (l == 1) ? W1 : (l == 2) ? W2 : W3;
    const int ps = pstart[l], pc = pcnt[l];
    v8f acc[4];
    #pragma unroll
    for (int nt = 0; nt < 4; ++nt) acc[nt] = (v8f){};
    for (int kc = 0; kc < 6; ++kc) {
      // protect previous chunk's readers before overwrite (also covers sZ/sF)
      __syncthreads();
      {
        // 128 threads x 16 floats = 2048 floats = rows k0..k0+31 of Wl
        const float* g = Wl + kc * 32 * 64 + tid * 16;
        const unsigned lds0 = (unsigned)(uintptr_t)(&sW[tid * 16]);
        #pragma unroll
        for (int q = 0; q < 4; ++q)
          async_g2l_b128(lds0 + q * 16u, g + q * 4);
        if (kc + 1 < 6)
          __builtin_prefetch(Wl + (kc + 1) * 32 * 64 + tid * 16, 0, 0);
        wait_asynccnt0();
      }
      __syncthreads();

      v16h a;
      #pragma unroll
      for (int j = 0; j < 16; ++j) {
        const int ka = kc * 32 + aKmap(j, h);
        a[j] = (ln < pc) ? sZ[w][(ps + ln) * 192 + ka] : (_Float16)0.0f;
      }
      #pragma unroll
      for (int nt = 0; nt < 4; ++nt) {
        v16h bb;
        #pragma unroll
        for (int j = 0; j < 16; ++j)
          bb[j] = (_Float16)sW[bKmap(j, h) * 64 + nt * 16 + ln];
        acc[nt] = wmma16x16x32(a, bb, acc[nt]);
      }
    }
    #pragma unroll
    for (int nt = 0; nt < 4; ++nt) {
      float ss = 0.0f;
      #pragma unroll
      for (int r = 0; r < 8; ++r) {
        const int row = r + 8 * h;
        if (row < pc) {
          float o = acc[nt][r];
          if (l == 0 && row == 0) o += b0[nt * 16 + ln];
          ss += o * o;
        }
      }
      ss += __shfl_down(ss, 16);                    // combine the two lane halves
      if (h == 0)
        h_all[(size_t)t * 256 + l * 64 + nt * 16 + ln] = sqrtf(fmaxf(ss, 1e-8f));
    }
  }
}

// ---------------------------------------------------------------------------
// Kernel 3: max over the M=512 targets per batch -> (B, 256)
// ---------------------------------------------------------------------------
__global__ void maxpool_kernel(const float* __restrict__ h_all,
                               float* __restrict__ hmax) {
  const int i = blockIdx.x * blockDim.x + threadIdx.x;   // 0..8191
  const int b = i >> 8, c = i & 255;
  float mx = -3.0e38f;
  for (int m = 0; m < M_; ++m)
    mx = fmaxf(mx, h_all[((size_t)b * M_ + m) * 256 + c]);
  hmax[i] = mx;
}

// ---------------------------------------------------------------------------
// Kernel 4: FC + ReLU (generic, tiny)
// ---------------------------------------------------------------------------
__global__ void fc_relu_kernel(const float* __restrict__ x,
                               const float* __restrict__ W,
                               const float* __restrict__ bias,
                               float* __restrict__ y, int In, int Out) {
  const int i = blockIdx.x * blockDim.x + threadIdx.x;
  if (i >= B_ * Out) return;
  const int b = i / Out, o = i - b * Out;
  float acc = bias[o];
  for (int k = 0; k < In; ++k) acc += x[b * In + k] * W[k * Out + o];
  y[i] = fmaxf(acc, 0.0f);
}

// ---------------------------------------------------------------------------
// Kernel 5: classifier head + softmax (one block per batch row)
// ---------------------------------------------------------------------------
__global__ void head_kernel(const float* __restrict__ h2,
                            const float* __restrict__ Wsm,
                            const float* __restrict__ bsm,
                            float* __restrict__ out) {
  __shared__ float sl[40];
  const int b = blockIdx.x;
  for (int o = threadIdx.x; o < 40; o += blockDim.x) {
    float acc = bsm[o];
    for (int k = 0; k < 256; ++k) acc += h2[b * 256 + k] * Wsm[k * 40 + o];
    sl[o] = acc;
  }
  __syncthreads();
  if (threadIdx.x == 0) {
    float mx = sl[0];
    for (int o = 1; o < 40; ++o) mx = fmaxf(mx, sl[o]);
    float sum = 0.0f;
    for (int o = 0; o < 40; ++o) { const float e = __expf(sl[o] - mx); sl[o] = e; sum += e; }
    const float inv = 1.0f / sum;
    for (int o = 0; o < 40; ++o) out[b * 40 + o] = sl[o] * inv;
  }
}

// ---------------------------------------------------------------------------
extern "C" void kernel_launch(void* const* d_in, const int* in_sizes, int n_in,
                              void* d_out, int out_size, void* d_ws, size_t ws_size,
                              hipStream_t stream) {
  const float* points = (const float*)d_in[0];
  const float* feats  = (const float*)d_in[1];
  const float* W0     = (const float*)d_in[2];
  const float* b0     = (const float*)d_in[3];
  const float* W1     = (const float*)d_in[4];
  const float* W2     = (const float*)d_in[5];
  const float* W3     = (const float*)d_in[6];
  const float* Wfc1   = (const float*)d_in[7];
  const float* bfc1   = (const float*)d_in[8];
  const float* Wfc2   = (const float*)d_in[9];
  const float* bfc2   = (const float*)d_in[10];
  const float* Wsm    = (const float*)d_in[11];
  const float* bsm    = (const float*)d_in[12];
  float* out = (float*)d_out;

  char* ws = (char*)d_ws;
  int*   idxbuf = (int*)  (ws);                         //  2 MB : 16384*32 int
  float* h_all  = (float*)(ws + 2097152);               // 16 MB : 16384*256 f32
  float* hmax   = (float*)(ws + 2097152 + 16777216);    // 32 KB : 32*256
  float* h1     = (float*)(ws + 2097152 + 16777216 + 32768);           // 64 KB
  float* h2     = (float*)(ws + 2097152 + 16777216 + 32768 + 65536);   // 32 KB

  // 16384 targets, 4 wave32-targets per 128-thread block
  knn_kernel<<<4096, 128, 0, stream>>>(points, idxbuf);
  tfn_kernel<<<4096, 128, 0, stream>>>(points, feats, idxbuf,
                                       W0, b0, W1, W2, W3, h_all);
  maxpool_kernel<<<32, 256, 0, stream>>>(h_all, hmax);               // 8192 thr
  fc_relu_kernel<<<64, 256, 0, stream>>>(hmax, Wfc1, bfc1, h1, 256, 512);
  fc_relu_kernel<<<32, 256, 0, stream>>>(h1, Wfc2, bfc2, h2, 512, 256);
  head_kernel<<<32, 64, 0, stream>>>(h2, Wsm, bsm, out);
}
